// MyMoEBlock_90185723281679
// MI455X (gfx1250) — matrix operand order repaired and checked by reading, not compiled, not run on hardware
//
#include <hip/hip_runtime.h>
#include <hip/hip_bf16.h>
#include <math.h>

// ---------------------------------------------------------------------------
// MoE block for MI455X (gfx1250, wave32, WMMA).
// router -> per-expert token grouping -> two grouped bf16 WMMA GEMMs
// (fp32 accumulate, KT=64, double-buffered LDS, global prefetch overlapped
// with WMMA stage) with GELU in between -> index-based deterministic combine.
// ---------------------------------------------------------------------------

#define D_MODEL 1024
#define HIDDEN  4096
#define NEXP    8
#define TOKENS  8192        // B*S = 4*2048
#define CAP     8192        // max tokens per expert (top-2 => <= TOKENS)
#define TOTAL_ROWS 16384    // TOKENS * top_k
#define KT      64          // K tile staged through LDS

typedef __bf16 bf16x8  __attribute__((ext_vector_type(8)));
typedef __bf16 bf16x16 __attribute__((ext_vector_type(16)));
typedef float  floatx8 __attribute__((ext_vector_type(8)));

static __device__ __forceinline__ bf16x16 cat8(bf16x8 lo, bf16x8 hi) {
  bf16x16 r;
#pragma unroll
  for (int i = 0; i < 8; ++i) { r[i] = lo[i]; r[i + 8] = hi[i]; }
  return r;
}

static __device__ __forceinline__ bf16x8 cvt8(float4 a, float4 b) {
  bf16x8 r;
  r[0] = (__bf16)a.x; r[1] = (__bf16)a.y; r[2] = (__bf16)a.z; r[3] = (__bf16)a.w;
  r[4] = (__bf16)b.x; r[5] = (__bf16)b.y; r[6] = (__bf16)b.z; r[7] = (__bf16)b.w;
  return r;
}

// ---------------------------------------------------------------------------
__global__ void k_zero_counts(int* counts) {
  if (threadIdx.x < NEXP) counts[threadIdx.x] = 0;
}

// ---------------------------------------------------------------------------
// Router: one wave32 per token: logits = x[t]·Wr, softmax, top-2.
// ---------------------------------------------------------------------------
__global__ __launch_bounds__(256) void k_router(
    const float* __restrict__ x, const float* __restrict__ Wr,
    int* __restrict__ counts, int* __restrict__ slot_e, int* __restrict__ slot_pos,
    int* __restrict__ tok_list, float* __restrict__ w_list) {
  const int wv   = threadIdx.x >> 5;
  const int lane = threadIdx.x & 31;
  const int t    = blockIdx.x * 8 + wv;

  float acc[NEXP];
#pragma unroll
  for (int e = 0; e < NEXP; ++e) acc[e] = 0.f;

  const float* xr = x + (size_t)t * D_MODEL;
  for (int k = lane; k < D_MODEL; k += 32) {
    float xv = xr[k];
    const float* wr = Wr + (size_t)k * NEXP;
#pragma unroll
    for (int e = 0; e < NEXP; ++e) acc[e] += xv * wr[e];
  }
#pragma unroll
  for (int e = 0; e < NEXP; ++e)
#pragma unroll
    for (int off = 16; off > 0; off >>= 1)
      acc[e] += __shfl_xor(acc[e], off, 32);

  if (lane == 0) {
    float mx = acc[0];
#pragma unroll
    for (int e = 1; e < NEXP; ++e) mx = fmaxf(mx, acc[e]);
    float p[NEXP], s = 0.f;
#pragma unroll
    for (int e = 0; e < NEXP; ++e) { p[e] = __expf(acc[e] - mx); s += p[e]; }
    float inv = 1.f / s;
    int e0 = 0; float v0 = p[0];
#pragma unroll
    for (int e = 1; e < NEXP; ++e) if (p[e] > v0) { v0 = p[e]; e0 = e; }
    int e1 = -1; float v1 = -1.f;
#pragma unroll
    for (int e = 0; e < NEXP; ++e) if (e != e0 && p[e] > v1) { v1 = p[e]; e1 = e; }
    v0 *= inv; v1 *= inv;

    int p0 = atomicAdd(&counts[e0], 1);
    tok_list[e0 * CAP + p0] = t; w_list[e0 * CAP + p0] = v0;
    slot_e[t * 2 + 0] = e0; slot_pos[t * 2 + 0] = p0;

    int p1 = atomicAdd(&counts[e1], 1);
    tok_list[e1 * CAP + p1] = t; w_list[e1 * CAP + p1] = v1;
    slot_e[t * 2 + 1] = e1; slot_pos[t * 2 + 1] = p1;
  }
}

// ---------------------------------------------------------------------------
__global__ void k_scan(const int* __restrict__ counts, int* __restrict__ offsets) {
  if (threadIdx.x == 0) {
    int s = 0;
#pragma unroll
    for (int e = 0; e < NEXP; ++e) { offsets[e] = s; s += counts[e]; }
  }
}

// ---------------------------------------------------------------------------
// Grouped GEMM1 + GELU.  h[row,:] = gelu(x[tok,:] @ W1[e] + b1[e])
// 128x128 tile, KT=64, double-buffered LDS, 8 waves x (2x4) WMMA frags.
// Loop order per iteration: ds_store(cur) -> barrier -> prefetch(next) ->
// WMMA(cur), so global fetch of tile j+1 overlaps the WMMA stage of tile j.
// ---------------------------------------------------------------------------
__global__ __launch_bounds__(256) void k_ffn1(
    const float* __restrict__ x, const float* __restrict__ W1,
    const float* __restrict__ b1,
    const int* __restrict__ counts, const int* __restrict__ offsets,
    const int* __restrict__ tok_list, __bf16* __restrict__ h) {
  const int e   = blockIdx.z;
  const int cnt = counts[e];
  const int m0  = blockIdx.x * 128;
  if (m0 >= cnt) return;                 // block-uniform exit
  const int n0  = blockIdx.y * 128;
  const int off = offsets[e];

  __shared__ __align__(16) __bf16 As[2][128][KT];
  __shared__ __align__(16) __bf16 Bs[2][128][KT];  // transposed: [n][k]
  __shared__ int toks[128];

  const int tid  = threadIdx.x;
  const int lane = tid & 31;
  const int w    = tid >> 5;
  const int waveM = w >> 1;              // 0..3 -> 32-row M strips
  const int waveN = w & 1;               // 0..1 -> 64-col N strips
  const int hi = lane >> 4, mn = lane & 15;

  if (tid < 128) {
    int gm = m0 + tid;
    toks[tid] = (gm < cnt) ? tok_list[e * CAP + gm] : -1;
  }
  __syncthreads();

  const float* W1e = W1 + (size_t)e * D_MODEL * HIDDEN;

  // staging ownership:
  //  A: thread = (row m = tid>>1, half = tid&1): 32 k's of one gathered row
  //  B: thread = (kg = tid>>5, np = tid&31): 8 k-rows x 4 n-cols sub-block
  const int sm = tid >> 1, shalf = tid & 1;
  const int kg = tid >> 5, np = tid & 31;
  const int stok = toks[sm];

  float4 vA[8], vB[8];
  auto loadA = [&](int k0) {
#pragma unroll
    for (int q = 0; q < 8; ++q) vA[q] = make_float4(0.f, 0.f, 0.f, 0.f);
    if (stok >= 0) {
      const float4* src = (const float4*)(x + (size_t)stok * D_MODEL + k0 + shalf * 32);
#pragma unroll
      for (int q = 0; q < 8; ++q) vA[q] = src[q];
    }
  };
  auto loadB = [&](int k0) {
    const float* base = W1e + (size_t)(k0 + kg * 8) * HIDDEN + n0 + np * 4;
#pragma unroll
    for (int r = 0; r < 8; ++r) vB[r] = *(const float4*)(base + (size_t)r * HIDDEN);
  };
  auto storeAB = [&](int buf) {
    bf16x8* dstA = (bf16x8*)&As[buf][sm][shalf * 32];
#pragma unroll
    for (int q = 0; q < 4; ++q) dstA[q] = cvt8(vA[2 * q], vA[2 * q + 1]);
#pragma unroll
    for (int j = 0; j < 4; ++j) {
      bf16x8 t;
#pragma unroll
      for (int r = 0; r < 8; ++r) t[r] = (__bf16)((const float*)&vB[r])[j];
      *(bf16x8*)&Bs[buf][np * 4 + j][kg * 8] = t;
    }
  };

  floatx8 acc[2][4];
#pragma unroll
  for (int fm = 0; fm < 2; ++fm)
#pragma unroll
    for (int fn = 0; fn < 4; ++fn) acc[fm][fn] = 0.f;

  loadA(0); loadB(0);
  const int NKT = D_MODEL / KT;
  for (int j = 0; j < NKT; ++j) {
    const int buf = j & 1;
    storeAB(buf);                          // ds_store current tile
    __syncthreads();                       // only DScnt must drain here
    if (j + 1 < NKT) {                     // prefetch next tile: overlaps WMMAs
      loadA((j + 1) * KT); loadB((j + 1) * KT);
    }
#pragma unroll
    for (int ks = 0; ks < KT; ks += 32) {
      const int c0 = ks >> 3;
      bf16x16 afrag[2];
#pragma unroll
      for (int fm = 0; fm < 2; ++fm) {
        const bf16x8* p = (const bf16x8*)&As[buf][waveM * 32 + fm * 16 + mn][0];
        afrag[fm] = cat8(p[c0 + hi], p[c0 + 2 + hi]);     // K: 8*hi.., 16+8*hi..
      }
#pragma unroll
      for (int fn = 0; fn < 4; ++fn) {
        const bf16x8* p = (const bf16x8*)&Bs[buf][waveN * 64 + fn * 16 + mn][0];
        bf16x16 bfrag = cat8(p[c0 + hi], p[c0 + 2 + hi]); // K: 8*hi.., 16+8*hi..
#pragma unroll
        for (int fm = 0; fm < 2; ++fm)
          acc[fm][fn] = __builtin_amdgcn_wmma_f32_16x16x32_bf16(
              false, afrag[fm], false, bfrag, (short)0, acc[fm][fn], false, false);
      }
    }
  }

  // epilogue: + b1, exact GELU, store bf16 h rows (compact layout)
#pragma unroll
  for (int fm = 0; fm < 2; ++fm)
#pragma unroll
    for (int fn = 0; fn < 4; ++fn) {
      int col = n0 + waveN * 64 + fn * 16 + mn;
      float bias = b1[e * HIDDEN + col];
#pragma unroll
      for (int v = 0; v < 8; ++v) {
        int mt = waveM * 32 + fm * 16 + v + 8 * hi;
        int gm = m0 + mt;
        if (gm < cnt) {
          float val = acc[fm][fn][v] + bias;
          float g = 0.5f * val * (1.f + erff(val * 0.70710678118654752f));
          h[(size_t)(off + gm) * HIDDEN + col] = (__bf16)g;
        }
      }
    }
}

// ---------------------------------------------------------------------------
// Grouped GEMM2.  y[row,:] = w_row * (h[row,:] @ W2[e] + b2[e])
// ---------------------------------------------------------------------------
__global__ __launch_bounds__(256) void k_ffn2(
    const __bf16* __restrict__ h, const float* __restrict__ W2,
    const float* __restrict__ b2,
    const int* __restrict__ counts, const int* __restrict__ offsets,
    const float* __restrict__ w_list, float* __restrict__ ybuf) {
  const int e   = blockIdx.z;
  const int cnt = counts[e];
  const int m0  = blockIdx.x * 128;
  if (m0 >= cnt) return;
  const int n0  = blockIdx.y * 128;
  const int off = offsets[e];

  __shared__ __align__(16) __bf16 As[2][128][KT];
  __shared__ __align__(16) __bf16 Bs[2][128][KT];

  const int tid  = threadIdx.x;
  const int lane = tid & 31;
  const int w    = tid >> 5;
  const int waveM = w >> 1, waveN = w & 1;
  const int hi = lane >> 4, mn = lane & 15;

  const float* W2e = W2 + (size_t)e * HIDDEN * D_MODEL;

  const int sm = tid >> 1, shalf = tid & 1;
  const int kg = tid >> 5, np = tid & 31;
  const bool arow = (m0 + sm) < cnt;
  const __bf16* hrow = h + (size_t)(off + m0 + sm) * HIDDEN + shalf * 32;

  bf16x8 vA[4];
  float4 vB[8];
  auto loadA = [&](int k0) {
    if (arow) {
      const bf16x8* src = (const bf16x8*)(hrow + k0);
#pragma unroll
      for (int q = 0; q < 4; ++q) vA[q] = src[q];
    } else {
      bf16x8 z;
#pragma unroll
      for (int i = 0; i < 8; ++i) z[i] = (__bf16)0.f;
#pragma unroll
      for (int q = 0; q < 4; ++q) vA[q] = z;
    }
  };
  auto loadB = [&](int k0) {
    const float* base = W2e + (size_t)(k0 + kg * 8) * D_MODEL + n0 + np * 4;
#pragma unroll
    for (int r = 0; r < 8; ++r) vB[r] = *(const float4*)(base + (size_t)r * D_MODEL);
  };
  auto storeAB = [&](int buf) {
    bf16x8* dstA = (bf16x8*)&As[buf][sm][shalf * 32];
#pragma unroll
    for (int q = 0; q < 4; ++q) dstA[q] = vA[q];
#pragma unroll
    for (int j = 0; j < 4; ++j) {
      bf16x8 t;
#pragma unroll
      for (int r = 0; r < 8; ++r) t[r] = (__bf16)((const float*)&vB[r])[j];
      *(bf16x8*)&Bs[buf][np * 4 + j][kg * 8] = t;
    }
  };

  floatx8 acc[2][4];
#pragma unroll
  for (int fm = 0; fm < 2; ++fm)
#pragma unroll
    for (int fn = 0; fn < 4; ++fn) acc[fm][fn] = 0.f;

  loadA(0); loadB(0);
  const int NKT = HIDDEN / KT;
  for (int j = 0; j < NKT; ++j) {
    const int buf = j & 1;
    storeAB(buf);
    __syncthreads();
    if (j + 1 < NKT) { loadA((j + 1) * KT); loadB((j + 1) * KT); }
#pragma unroll
    for (int ks = 0; ks < KT; ks += 32) {
      const int c0 = ks >> 3;
      bf16x16 afrag[2];
#pragma unroll
      for (int fm = 0; fm < 2; ++fm) {
        const bf16x8* p = (const bf16x8*)&As[buf][waveM * 32 + fm * 16 + mn][0];
        afrag[fm] = cat8(p[c0 + hi], p[c0 + 2 + hi]);
      }
#pragma unroll
      for (int fn = 0; fn < 4; ++fn) {
        const bf16x8* p = (const bf16x8*)&Bs[buf][waveN * 64 + fn * 16 + mn][0];
        bf16x16 bfrag = cat8(p[c0 + hi], p[c0 + 2 + hi]);
#pragma unroll
        for (int fm = 0; fm < 2; ++fm)
          acc[fm][fn] = __builtin_amdgcn_wmma_f32_16x16x32_bf16(
              false, afrag[fm], false, bfrag, (short)0, acc[fm][fn], false, false);
      }
    }
  }

#pragma unroll
  for (int fm = 0; fm < 2; ++fm)
#pragma unroll
    for (int fn = 0; fn < 4; ++fn) {
      int col = n0 + waveN * 64 + fn * 16 + mn;
      float bias = b2[e * D_MODEL + col];
#pragma unroll
      for (int v = 0; v < 8; ++v) {
        int mt = waveM * 32 + fm * 16 + v + 8 * hi;
        int gm = m0 + mt;
        if (gm < cnt) {
          float wgt = w_list[e * CAP + gm];
          ybuf[(size_t)(off + gm) * D_MODEL + col] = wgt * (acc[fm][fn][v] + bias);
        }
      }
    }
}

// ---------------------------------------------------------------------------
// Combine. out[t,c] = y[row(t,0),c] + y[row(t,1),c].
// ---------------------------------------------------------------------------
__global__ __launch_bounds__(256) void k_combine(
    const float* __restrict__ ybuf, const int* __restrict__ slot_e,
    const int* __restrict__ slot_pos, const int* __restrict__ offsets,
    float* __restrict__ out) {
  int i = blockIdx.x * 256 + threadIdx.x;        // T*D = 8388608, exact grid
  int t = i >> 10;                                // D_MODEL = 1024
  int c = i & 1023;
  int e0 = slot_e[t * 2 + 0], p0 = slot_pos[t * 2 + 0];
  int e1 = slot_e[t * 2 + 1], p1 = slot_pos[t * 2 + 1];
  out[i] = ybuf[(size_t)(offsets[e0] + p0) * D_MODEL + c] +
           ybuf[(size_t)(offsets[e1] + p1) * D_MODEL + c];
}

// ---------------------------------------------------------------------------
extern "C" void kernel_launch(void* const* d_in, const int* in_sizes, int n_in,
                              void* d_out, int out_size, void* d_ws, size_t ws_size,
                              hipStream_t stream) {
  const float* x  = (const float*)d_in[0];
  const float* Wr = (const float*)d_in[1];
  const float* W1 = (const float*)d_in[2];
  const float* b1 = (const float*)d_in[3];
  const float* W2 = (const float*)d_in[4];
  const float* b2 = (const float*)d_in[5];
  float* out = (float*)d_out;

  // Workspace layout (~203 MB total).
  char* ws = (char*)d_ws;
  size_t off = 0;
  int*   counts   = (int*)(ws + off);  off += 256;
  int*   offsets  = (int*)(ws + off);  off += 256;
  int*   slot_e   = (int*)(ws + off);  off += (size_t)TOKENS * 2 * sizeof(int);
  int*   slot_pos = (int*)(ws + off);  off += (size_t)TOKENS * 2 * sizeof(int);
  int*   tok_list = (int*)(ws + off);  off += (size_t)NEXP * CAP * sizeof(int);
  float* w_list   = (float*)(ws + off); off += (size_t)NEXP * CAP * sizeof(float);
  off = (off + 255) & ~(size_t)255;
  __bf16* hbuf = (__bf16*)(ws + off);  off += (size_t)TOTAL_ROWS * HIDDEN * sizeof(__bf16);
  float*  ybuf = (float*)(ws + off);   off += (size_t)TOTAL_ROWS * D_MODEL * sizeof(float);

  k_zero_counts<<<1, 32, 0, stream>>>(counts);
  k_router<<<TOKENS / 8, 256, 0, stream>>>(x, Wr, counts, slot_e, slot_pos,
                                           tok_list, w_list);
  k_scan<<<1, 1, 0, stream>>>(counts, offsets);
  k_ffn1<<<dim3(CAP / 128, HIDDEN / 128, NEXP), 256, 0, stream>>>(
      x, W1, b1, counts, offsets, tok_list, hbuf);
  k_ffn2<<<dim3(CAP / 128, D_MODEL / 128, NEXP), 256, 0, stream>>>(
      hbuf, W2, b2, counts, offsets, w_list, ybuf);
  k_combine<<<(TOKENS * D_MODEL) / 256, 256, 0, stream>>>(ybuf, slot_e, slot_pos,
                                                          offsets, out);
}
